// MolEncoder_22050362098317
// MI455X (gfx1250) — compile-verified, matrix operand order checked by reading, not run on hardware
//
#include <hip/hip_runtime.h>
#include <hip/hip_bf16.h>
#include <math.h>

// ---------------------------------------------------------------------------
// MolEncoder (EGNN-style) for gfx1250: f16 WMMA GEMMs with 32-row M-tiles,
// float4-vectorized LDS staging, atomic segment-sums. Wave32 / WGP per CDNA5.
// ---------------------------------------------------------------------------

typedef __attribute__((ext_vector_type(16))) _Float16 v16h;
typedef __attribute__((ext_vector_type(4)))  _Float16 v4h;
typedef __attribute__((ext_vector_type(8)))  float    v8f;
typedef __attribute__((ext_vector_type(4)))  float    v4f;

#define C_NN   128   // node feature width
#define C_NE   64    // edge feature width
#define C_H    256   // hidden width
#define C_OUT  128   // node_out width
#define C_EMB  64    // embedding width

#define KPAD_E 352   // 321 -> 11 k-tiles of 32
#define KT_E   11
#define KPAD_N 224   // 195 -> 7 k-tiles of 32
#define KT_N   7

#define MT     32    // rows (edges/nodes) per workgroup: two 16-row WMMA tiles

union FragH { unsigned u[8]; v16h h; };

// --- WMMA wrapper: D = A(16x32 f16) * B(32x16 f16) + C(16x16 f32) -----------
__device__ __forceinline__ v8f wmma_f16(v16h a, v16h b, v8f c) {
  return __builtin_amdgcn_wmma_f32_16x16x32_f16(
      /*neg_a=*/false, a, /*neg_b=*/false, b,
      /*c_mod=*/(short)0, c, /*reuse_a=*/false, /*reuse_b=*/false);
}

// K index for VGPR v of an A/B fragment (CDNA5 ISA 7.12.2, 16-bit A 16x32):
// lanes 0-15: v0..3 -> K=0..7, v4..7 -> K=16..23 ; lanes 16-31 shifted +8.
__device__ __forceinline__ int frag_k0(int v, int half) {
  return (v < 4) ? (v * 2 + half * 8) : (16 + (v - 4) * 2 + half * 8);
}

// Load A fragment from an LDS tile stored row-major [16][kpad] f16.
__device__ __forceinline__ v16h load_a_frag(const _Float16* lds, int kpad,
                                            int kt, int lane) {
  const unsigned* p = (const unsigned*)lds;   // packed f16 pairs
  int half = lane >> 4, m = lane & 15;
  int rowbase = m * (kpad >> 1);
  FragH f;
#pragma unroll
  for (int v = 0; v < 8; ++v) {
    int k0 = kt * 32 + frag_k0(v, half);
    f.u[v] = p[rowbase + (k0 >> 1)];
  }
  return f.h;
}

// Load B fragment from pre-packed weights (8 contiguous dwords per lane).
__device__ __forceinline__ v16h load_b_frag(const unsigned* __restrict__ wp,
                                            int nt_count, int kt, int nt,
                                            int lane) {
  const unsigned* src = wp + ((size_t)((kt * nt_count) + nt) * 32 + lane) * 8;
  FragH f;
#pragma unroll
  for (int v = 0; v < 8; ++v) f.u[v] = src[v];
  return f.h;
}

__device__ __forceinline__ v4h cvt4(v4f x) {
  v4h h;
#pragma unroll
  for (int k = 0; k < 4; ++k) h[k] = (_Float16)x[k];
  return h;
}

// --- Weight pre-pack: f32 row-major [K][Nn] -> B-fragment layout ------------
__global__ void pack_b_kernel(const float* __restrict__ W,
                              unsigned* __restrict__ dst,
                              int K, int Nn, int KT, int NT) {
  int t = blockIdx.x * blockDim.x + threadIdx.x;
  int total = KT * NT * 32;
  if (t >= total) return;
  int lane = t & 31;
  int ft = t >> 5;
  int nt = ft % NT, kt = ft / NT;
  int half = lane >> 4, n = lane & 15;
  int col = nt * 16 + n;
  unsigned* d = dst + (size_t)t * 8;
#pragma unroll
  for (int v = 0; v < 8; ++v) {
    int k0 = kt * 32 + frag_k0(v, half);
    float a = (k0     < K) ? W[(size_t)k0 * Nn + col]       : 0.f;
    float b = (k0 + 1 < K) ? W[(size_t)(k0 + 1) * Nn + col] : 0.f;
    _Float16 ah = (_Float16)a, bh = (_Float16)b;
    unsigned short au = __builtin_bit_cast(unsigned short, ah);
    unsigned short bu = __builtin_bit_cast(unsigned short, bh);
    d[v] = (unsigned)au | ((unsigned)bu << 16);
  }
}

// ---------------------------------------------------------------------------
// Edge pipeline: one workgroup (4 wave32) handles a 32-edge tile (2 M-tiles).
//   edge_in = [nf[ai] | nf[aj] | edge_attr | radial]  (K=321 -> 352)
//   hidden  = relu(edge_in @ W1 + b1)    (32x256)
//   eout    = hidden @ W2 + b2           (32x64)
//   e_c     = relu(eout @ ce_w1 + ce_b1) @ ce_w2       (scalar per edge)
//   atomics: coord_accum[ai] += normed * e_c ; agg_e[ai] += eout
//   edge_emb = eout @ eemb_w + eemb_b -> d_out
// ---------------------------------------------------------------------------
__global__ __launch_bounds__(128)
void edge_kernel(const float* __restrict__ node_feats,
                 const int*   __restrict__ edge_index,
                 const float* __restrict__ edge_attr,
                 const float* __restrict__ coords,
                 const unsigned* __restrict__ W1p, const float* __restrict__ b1,
                 const unsigned* __restrict__ W2p, const float* __restrict__ b2,
                 const float* __restrict__ ce_w1, const float* __restrict__ ce_b1,
                 const float* __restrict__ ce_w2,
                 const unsigned* __restrict__ Wep, const float* __restrict__ eemb_b,
                 float* __restrict__ agg_e, float* __restrict__ coord_accum,
                 float* __restrict__ edge_emb_out,
                 long E) {
  __shared__ _Float16 sA[MT * KPAD_E];     // edge_in tile (f16)
  __shared__ _Float16 sH[MT * C_H];        // hidden tile  (f16)
  __shared__ float    sOut[MT * C_NE];     // edge_out tile (f32)
  __shared__ _Float16 sOutH[MT * C_NE];    // edge_out tile (f16)
  __shared__ float    sNormed[MT * 3];
  __shared__ float    sRadial[MT];
  __shared__ int      sAi[MT];
  __shared__ int      sAj[MT];

  const int tid  = threadIdx.x;
  const int lane = tid & 31;
  const int wave = tid >> 5;
  const int half = lane >> 4;
  const int n    = lane & 15;
  const long e0  = (long)blockIdx.x * MT;

  // per-edge geometry (radial, normed), index caching
  if (tid < MT) {
    long e = e0 + tid;
    if (e < E) {
      int ai = edge_index[e];
      int aj = edge_index[E + e];
      sAi[tid] = ai; sAj[tid] = aj;
      float dx = coords[(size_t)ai * 3 + 0] - coords[(size_t)aj * 3 + 0];
      float dy = coords[(size_t)ai * 3 + 1] - coords[(size_t)aj * 3 + 1];
      float dz = coords[(size_t)ai * 3 + 2] - coords[(size_t)aj * 3 + 2];
      float r  = dx * dx + dy * dy + dz * dz;
      sRadial[tid] = r;
      float inv = 1.0f / (sqrtf(r) + 1.0f);
      sNormed[tid * 3 + 0] = dx * inv;
      sNormed[tid * 3 + 1] = dy * inv;
      sNormed[tid * 3 + 2] = dz * inv;
    } else {
      sAi[tid] = 0; sAj[tid] = 0; sRadial[tid] = 0.f;
      sNormed[tid * 3 + 0] = 0.f; sNormed[tid * 3 + 1] = 0.f;
      sNormed[tid * 3 + 2] = 0.f;
    }
  }
  __syncthreads();

  // stage concatenated edge_in tile into LDS (f16), float4-vectorized
  for (int i = tid; i < MT * (KPAD_E / 4); i += 128) {
    int el = i / (KPAD_E / 4);
    int c4 = (i % (KPAD_E / 4)) * 4;
    long e = e0 + el;
    v4f val = {0.f, 0.f, 0.f, 0.f};
    if (e < E) {
      if (c4 < 128)
        val = *(const v4f*)(node_feats + (size_t)sAi[el] * C_NN + c4);
      else if (c4 < 256)
        val = *(const v4f*)(node_feats + (size_t)sAj[el] * C_NN + (c4 - 128));
      else if (c4 < 320)
        val = *(const v4f*)(edge_attr + (size_t)e * C_NE + (c4 - 256));
      else if (c4 == 320)
        val = (v4f){sRadial[el], 0.f, 0.f, 0.f};
    }
    *(v4h*)(sA + el * KPAD_E + c4) = cvt4(val);
  }
  __syncthreads();

  // ---- GEMM 1: [32 x 352] x [352 x 256], 4 n-tiles x 2 m-tiles per wave ----
  v8f acc[2][4] = {};
#pragma unroll 1
  for (int kt = 0; kt < KT_E; ++kt) {
    v16h a0 = load_a_frag(sA,                 KPAD_E, kt, lane);
    v16h a1 = load_a_frag(sA + 16 * KPAD_E,   KPAD_E, kt, lane);
#pragma unroll
    for (int j = 0; j < 4; ++j) {
      v16h b = load_b_frag(W1p, 16, kt, wave * 4 + j, lane);
      acc[0][j] = wmma_f16(a0, b, acc[0][j]);
      acc[1][j] = wmma_f16(a1, b, acc[1][j]);
    }
  }
#pragma unroll
  for (int j = 0; j < 4; ++j) {
    int col = (wave * 4 + j) * 16 + n;
    float bb = b1[col];
#pragma unroll
    for (int mt = 0; mt < 2; ++mt) {
#pragma unroll
      for (int v = 0; v < 8; ++v) {
        int m = mt * 16 + v + half * 8;
        float x = acc[mt][j][v] + bb;
        sH[m * C_H + col] = (_Float16)(x > 0.f ? x : 0.f);
      }
    }
  }
  __syncthreads();

  // ---- GEMM 2: [32 x 256] x [256 x 64], 1 n-tile x 2 m-tiles per wave ----
  v8f acc2[2] = {};
#pragma unroll 1
  for (int kt = 0; kt < 8; ++kt) {
    v16h a0 = load_a_frag(sH,              C_H, kt, lane);
    v16h a1 = load_a_frag(sH + 16 * C_H,   C_H, kt, lane);
    v16h b  = load_b_frag(W2p, 4, kt, wave, lane);
    acc2[0] = wmma_f16(a0, b, acc2[0]);
    acc2[1] = wmma_f16(a1, b, acc2[1]);
  }
  {
    int col = wave * 16 + n;
    float bb = b2[col];
#pragma unroll
    for (int mt = 0; mt < 2; ++mt) {
#pragma unroll
      for (int v = 0; v < 8; ++v) {
        int m = mt * 16 + v + half * 8;
        sOut[m * C_NE + col] = acc2[mt][v] + bb;
      }
    }
  }
  __syncthreads();

  // ---- tiny coord MLP (64->2->1) + trans scatter-add ----
  if (tid < MT) {
    long e = e0 + tid;
    if (e < E) {
      float h0 = ce_b1[0], h1 = ce_b1[1];
#pragma unroll 4
      for (int k = 0; k < C_NE; ++k) {
        float x = sOut[tid * C_NE + k];
        h0 += x * ce_w1[k * 2 + 0];
        h1 += x * ce_w1[k * 2 + 1];
      }
      h0 = h0 > 0.f ? h0 : 0.f;
      h1 = h1 > 0.f ? h1 : 0.f;
      float ec = h0 * ce_w2[0] + h1 * ce_w2[1];
      int ai = sAi[tid];
      atomicAdd(&coord_accum[(size_t)ai * 3 + 0], sNormed[tid * 3 + 0] * ec);
      atomicAdd(&coord_accum[(size_t)ai * 3 + 1], sNormed[tid * 3 + 1] * ec);
      atomicAdd(&coord_accum[(size_t)ai * 3 + 2], sNormed[tid * 3 + 2] * ec);
    }
  }

  // ---- agg_e scatter-add + f16 copy of eout for embedding GEMM ----
  for (int i = tid; i < MT * C_NE; i += 128) {
    int el = i >> 6, col = i & 63;
    float x = sOut[i];
    sOutH[i] = (_Float16)x;
    long e = e0 + el;
    if (e < E) atomicAdd(&agg_e[(size_t)sAi[el] * C_NE + col], x);
  }
  __syncthreads();

  // ---- GEMM 3: edge_emb = [32 x 64] x [64 x 64], 1 n-tile x 2 m-tiles ----
  v8f acc3[2] = {};
#pragma unroll
  for (int kt = 0; kt < 2; ++kt) {
    v16h a0 = load_a_frag(sOutH,             C_NE, kt, lane);
    v16h a1 = load_a_frag(sOutH + 16 * C_NE, C_NE, kt, lane);
    v16h b  = load_b_frag(Wep, 4, kt, wave, lane);
    acc3[0] = wmma_f16(a0, b, acc3[0]);
    acc3[1] = wmma_f16(a1, b, acc3[1]);
  }
  {
    int col = wave * 16 + n;
    float bb = eemb_b[col];
#pragma unroll
    for (int mt = 0; mt < 2; ++mt) {
#pragma unroll
      for (int v = 0; v < 8; ++v) {
        int m = mt * 16 + v + half * 8;
        long e = e0 + m;
        if (e < E) edge_emb_out[(size_t)e * C_EMB + col] = acc3[mt][v] + bb;
      }
    }
  }
}

// ---------------------------------------------------------------------------
// coord_out = (coords + coord_accum) @ coord_w + coord_b ; counts[batch] += 1
// ---------------------------------------------------------------------------
__global__ __launch_bounds__(256)
void coord_kernel(const float* __restrict__ coords,
                  const float* __restrict__ coord_accum,
                  const float* __restrict__ coord_w,
                  const float* __restrict__ coord_b,
                  const int*   __restrict__ batch,
                  float* __restrict__ counts,
                  float* __restrict__ coord_out, int Nn) {
  int node = blockIdx.x * blockDim.x + threadIdx.x;
  if (node >= Nn) return;
  float c[3];
#pragma unroll
  for (int d = 0; d < 3; ++d)
    c[d] = coords[(size_t)node * 3 + d] + coord_accum[(size_t)node * 3 + d];
#pragma unroll
  for (int d = 0; d < 3; ++d) {
    float o = coord_b[d];
#pragma unroll
    for (int e = 0; e < 3; ++e) o += c[e] * coord_w[e * 3 + d];
    coord_out[(size_t)node * 3 + d] = o;
  }
  atomicAdd(&counts[batch[node]], 1.0f);
}

// ---------------------------------------------------------------------------
// Node pipeline: 32-node tile per workgroup (2 M-tiles).
//   node_in = [node_feats | agg_e | coord_out]  (K=195 -> 224)
//   hidden  = relu(node_in @ W1 + b1)   (32x256)
//   nout    = hidden @ W2 + b2          (32x128, linear)
//   node_emb = nout @ nemb_w + nemb_b -> d_out ; sums[batch] += node_emb
// ---------------------------------------------------------------------------
__global__ __launch_bounds__(128)
void node_kernel(const float* __restrict__ node_feats,
                 const float* __restrict__ agg_e,
                 const float* __restrict__ coord_out,
                 const unsigned* __restrict__ W1p, const float* __restrict__ b1,
                 const unsigned* __restrict__ W2p, const float* __restrict__ b2,
                 const unsigned* __restrict__ Wnp, const float* __restrict__ nemb_b,
                 const int* __restrict__ batch,
                 float* __restrict__ node_emb_out, float* __restrict__ sums,
                 int Nn) {
  __shared__ _Float16 sA[MT * KPAD_N];
  __shared__ _Float16 sH[MT * C_H];
  __shared__ _Float16 sNO[MT * C_OUT];
  __shared__ int      sB[MT];

  const int tid  = threadIdx.x;
  const int lane = tid & 31;
  const int wave = tid >> 5;
  const int half = lane >> 4;
  const int n    = lane & 15;
  const long n0  = (long)blockIdx.x * MT;

  if (tid < MT) {
    long nd = n0 + tid;
    sB[tid] = (nd < Nn) ? batch[nd] : 0;
  }

  // stage node_in tile (f16), float4-vectorized; special chunk at col 192
  for (int i = tid; i < MT * (KPAD_N / 4); i += 128) {
    int nl = i / (KPAD_N / 4);
    int c4 = (i % (KPAD_N / 4)) * 4;
    long nd = n0 + nl;
    v4f val = {0.f, 0.f, 0.f, 0.f};
    if (nd < Nn) {
      if (c4 < 128)
        val = *(const v4f*)(node_feats + (size_t)nd * C_NN + c4);
      else if (c4 < 192)
        val = *(const v4f*)(agg_e + (size_t)nd * C_NE + (c4 - 128));
      else if (c4 == 192) {
        val[0] = coord_out[(size_t)nd * 3 + 0];
        val[1] = coord_out[(size_t)nd * 3 + 1];
        val[2] = coord_out[(size_t)nd * 3 + 2];
      }
    }
    *(v4h*)(sA + nl * KPAD_N + c4) = cvt4(val);
  }
  __syncthreads();

  // ---- GEMM 1: [32 x 224] x [224 x 256], 4 n-tiles x 2 m-tiles per wave ----
  v8f acc[2][4] = {};
#pragma unroll 1
  for (int kt = 0; kt < KT_N; ++kt) {
    v16h a0 = load_a_frag(sA,               KPAD_N, kt, lane);
    v16h a1 = load_a_frag(sA + 16 * KPAD_N, KPAD_N, kt, lane);
#pragma unroll
    for (int j = 0; j < 4; ++j) {
      v16h b = load_b_frag(W1p, 16, kt, wave * 4 + j, lane);
      acc[0][j] = wmma_f16(a0, b, acc[0][j]);
      acc[1][j] = wmma_f16(a1, b, acc[1][j]);
    }
  }
#pragma unroll
  for (int j = 0; j < 4; ++j) {
    int col = (wave * 4 + j) * 16 + n;
    float bb = b1[col];
#pragma unroll
    for (int mt = 0; mt < 2; ++mt) {
#pragma unroll
      for (int v = 0; v < 8; ++v) {
        int m = mt * 16 + v + half * 8;
        float x = acc[mt][j][v] + bb;
        sH[m * C_H + col] = (_Float16)(x > 0.f ? x : 0.f);
      }
    }
  }
  __syncthreads();

  // ---- GEMM 2: [32 x 256] x [256 x 128], 2 n-tiles x 2 m-tiles (linear) ----
  v8f acc2[2][2] = {};
#pragma unroll 1
  for (int kt = 0; kt < 8; ++kt) {
    v16h a0 = load_a_frag(sH,            C_H, kt, lane);
    v16h a1 = load_a_frag(sH + 16 * C_H, C_H, kt, lane);
#pragma unroll
    for (int j = 0; j < 2; ++j) {
      v16h b = load_b_frag(W2p, 8, kt, wave * 2 + j, lane);
      acc2[0][j] = wmma_f16(a0, b, acc2[0][j]);
      acc2[1][j] = wmma_f16(a1, b, acc2[1][j]);
    }
  }
#pragma unroll
  for (int j = 0; j < 2; ++j) {
    int col = (wave * 2 + j) * 16 + n;
    float bb = b2[col];
#pragma unroll
    for (int mt = 0; mt < 2; ++mt) {
#pragma unroll
      for (int v = 0; v < 8; ++v) {
        int m = mt * 16 + v + half * 8;
        sNO[m * C_OUT + col] = (_Float16)(acc2[mt][j][v] + bb);
      }
    }
  }
  __syncthreads();

  // ---- GEMM 3: node_emb = [32 x 128] x [128 x 64], 1 n-tile x 2 m-tiles ----
  v8f acc3[2] = {};
#pragma unroll
  for (int kt = 0; kt < 4; ++kt) {
    v16h a0 = load_a_frag(sNO,              C_OUT, kt, lane);
    v16h a1 = load_a_frag(sNO + 16 * C_OUT, C_OUT, kt, lane);
    v16h b  = load_b_frag(Wnp, 4, kt, wave, lane);
    acc3[0] = wmma_f16(a0, b, acc3[0]);
    acc3[1] = wmma_f16(a1, b, acc3[1]);
  }
  {
    int col = wave * 16 + n;
    float bb = nemb_b[col];
#pragma unroll
    for (int mt = 0; mt < 2; ++mt) {
#pragma unroll
      for (int v = 0; v < 8; ++v) {
        int m = mt * 16 + v + half * 8;
        long nd = n0 + m;
        if (nd < Nn) {
          float x = acc3[mt][v] + bb;
          node_emb_out[(size_t)nd * C_EMB + col] = x;
          atomicAdd(&sums[(size_t)sB[m] * C_EMB + col], x);
        }
      }
    }
  }
}

// ---------------------------------------------------------------------------
// graph_emb = sums / max(counts, 1)
// ---------------------------------------------------------------------------
__global__ __launch_bounds__(256)
void graph_kernel(const float* __restrict__ sums, const float* __restrict__ counts,
                  float* __restrict__ graph_emb, int total) {
  int i = blockIdx.x * blockDim.x + threadIdx.x;
  if (i >= total) return;
  float c = counts[i >> 6];
  graph_emb[i] = sums[i] / (c > 1.0f ? c : 1.0f);
}

// ---------------------------------------------------------------------------
extern "C" void kernel_launch(void* const* d_in, const int* in_sizes, int n_in,
                              void* d_out, int out_size, void* d_ws, size_t ws_size,
                              hipStream_t stream) {
  const float* node_feats = (const float*)d_in[0];
  const int*   edge_index = (const int*)d_in[1];
  const float* edge_attr  = (const float*)d_in[2];
  const float* coords     = (const float*)d_in[3];
  const int*   batch      = (const int*)d_in[4];
  const float* edge_w1 = (const float*)d_in[6];
  const float* edge_b1 = (const float*)d_in[7];
  const float* edge_w2 = (const float*)d_in[8];
  const float* edge_b2 = (const float*)d_in[9];
  const float* node_w1 = (const float*)d_in[10];
  const float* node_b1 = (const float*)d_in[11];
  const float* node_w2 = (const float*)d_in[12];
  const float* node_b2 = (const float*)d_in[13];
  const float* ce_w1   = (const float*)d_in[14];
  const float* ce_b1   = (const float*)d_in[15];
  const float* ce_w2   = (const float*)d_in[16];
  const float* coord_w = (const float*)d_in[17];
  const float* coord_b = (const float*)d_in[18];
  const float* nemb_w  = (const float*)d_in[19];
  const float* nemb_b  = (const float*)d_in[20];
  const float* eemb_w  = (const float*)d_in[21];
  const float* eemb_b  = (const float*)d_in[22];

  const int  Nn = in_sizes[0] / C_NN;           // nodes
  const long E  = (long)in_sizes[1] / 2;        // edges
  // G derived from out_size: out = N*EMB + E*EMB + G*EMB + N*3
  const long fixed = (long)Nn * C_EMB + E * C_EMB + (long)Nn * 3;
  const int  G = (int)(((long)out_size - fixed) / C_EMB);

  // output layout
  float* out_node_emb  = (float*)d_out;
  float* out_edge_emb  = out_node_emb + (size_t)Nn * C_EMB;
  float* out_graph_emb = out_edge_emb + (size_t)E * C_EMB;
  float* out_coord     = out_graph_emb + (size_t)G * C_EMB;

  // workspace layout (zero-init region first, then packed weights)
  char* ws = (char*)d_ws;
  auto align256 = [](size_t x) { return (x + 255) & ~(size_t)255; };
  size_t off = 0;
  float* agg_e       = (float*)(ws + off); off = align256(off + (size_t)Nn * C_NE * 4);
  float* coord_accum = (float*)(ws + off); off = align256(off + (size_t)Nn * 3 * 4);
  float* sums        = (float*)(ws + off); off = align256(off + (size_t)G * C_EMB * 4);
  float* counts      = (float*)(ws + off); off = align256(off + (size_t)G * 4);
  size_t zero_bytes = off;
  unsigned* W1e = (unsigned*)(ws + off); off = align256(off + (size_t)KT_E * 16 * 256 * 4);
  unsigned* W2e = (unsigned*)(ws + off); off = align256(off + (size_t)8 * 4 * 256 * 4);
  unsigned* W1n = (unsigned*)(ws + off); off = align256(off + (size_t)KT_N * 16 * 256 * 4);
  unsigned* W2n = (unsigned*)(ws + off); off = align256(off + (size_t)8 * 8 * 256 * 4);
  unsigned* Wne = (unsigned*)(ws + off); off = align256(off + (size_t)4 * 4 * 256 * 4);
  unsigned* Wee = (unsigned*)(ws + off); off = align256(off + (size_t)2 * 4 * 256 * 4);
  (void)ws_size; (void)n_in;

  hipMemsetAsync(d_ws, 0, zero_bytes, stream);

  // pack all WMMA weight matrices into B-fragment layout
  auto pack = [&](const float* W, unsigned* dst, int K, int Nc, int KT, int NT) {
    int total = KT * NT * 32;
    pack_b_kernel<<<(total + 255) / 256, 256, 0, stream>>>(W, dst, K, Nc, KT, NT);
  };
  pack(edge_w1, W1e, 321, 256, KT_E, 16);
  pack(edge_w2, W2e, 256, 64, 8, 4);
  pack(node_w1, W1n, 195, 256, KT_N, 16);
  pack(node_w2, W2n, 256, 128, 8, 8);
  pack(nemb_w,  Wne, 128, 64, 4, 4);
  pack(eemb_w,  Wee, 64, 64, 2, 4);

  // edge pipeline
  {
    long blocks = (E + MT - 1) / MT;
    edge_kernel<<<dim3((unsigned)blocks), 128, 0, stream>>>(
        node_feats, edge_index, edge_attr, coords,
        W1e, edge_b1, W2e, edge_b2, ce_w1, ce_b1, ce_w2, Wee, eemb_b,
        agg_e, coord_accum, out_edge_emb, E);
  }

  // coord_out + graph counts
  coord_kernel<<<(Nn + 255) / 256, 256, 0, stream>>>(
      coords, coord_accum, coord_w, coord_b, batch, counts, out_coord, Nn);

  // node pipeline
  {
    int blocks = (Nn + MT - 1) / MT;
    node_kernel<<<blocks, 128, 0, stream>>>(
        node_feats, agg_e, out_coord,
        W1n, node_b1, W2n, node_b2, Wne, nemb_b,
        batch, out_node_emb, sums, Nn);
  }

  // graph mean pool
  {
    int total = G * C_EMB;
    graph_kernel<<<(total + 255) / 256, 256, 0, stream>>>(
        sums, counts, out_graph_emb, total);
  }
}
